// GPT_66331474920219
// MI455X (gfx1250) — matrix operand order repaired and checked by reading, not compiled, not run on hardware
//
#include <hip/hip_runtime.h>
#include <math.h>
#include <stdint.h>

// ---------------- problem constants (match reference) ----------------
#define Vv   32000
#define Dd   1024
#define Hh   16
#define Ll   6
#define Tt   1024
#define Bb   2
#define HD   64           // Dd / Hh
#define MTOK (Bb * Tt)    // 2048 rows of activations

typedef unsigned short u16;
typedef __attribute__((ext_vector_type(16))) __bf16          bvec16;
typedef __attribute__((ext_vector_type(8)))  float           fvec8;
typedef __attribute__((ext_vector_type(8)))  unsigned short  usvec8;
typedef __attribute__((ext_vector_type(4)))  unsigned int    uivec4;
typedef __attribute__((ext_vector_type(8)))  int             ivec8;
typedef __attribute__((ext_vector_type(4)))  int             ivec4;

// A/B operand fragment for v_wmma_f32_16x16x32_bf16: 16 bf16 = 2 x 16B chunks.
union AFrag {
  bvec16 v;
  usvec8 h[2];
};

__device__ __forceinline__ u16 f2bf(float f) {
  unsigned u = __float_as_uint(f);
  u += 0x7FFFu + ((u >> 16) & 1u);   // round-to-nearest-even
  return (u16)(u >> 16);
}

__device__ __forceinline__ fvec8 zero8() {
  fvec8 z = {0.f, 0.f, 0.f, 0.f, 0.f, 0.f, 0.f, 0.f};
  return z;
}

#if defined(__has_builtin)
#if __has_builtin(__builtin_amdgcn_tensor_load_to_lds) && __has_builtin(__builtin_amdgcn_s_wait_tensorcnt)
#define USE_TDM 1
#endif
#endif

#ifdef USE_TDM
// Issue one TDM 2-D tile load: rows x rowlen bf16 elements, row stride = stride
// elements, packed row-major into LDS at lds_off. D# bit layout per CDNA5 ISA
// ch.8 (group0: flags/lds/global addr/type=2; group1: data_size=2B, dims/strides).
// This toolchain exposes the 6-arg builtin: (v4u g0, v8i g1, v4i g2, v4i g3, v8i, i32 cpol).
__device__ __forceinline__ void tdm_load_2d(unsigned lds_off, const u16* gptr,
                                            int rows, int rowlen, int stride) {
  unsigned long long ga = (unsigned long long)(size_t)gptr;
  uivec4 g0;
  g0[0] = 1u;                                          // count=1, user desc
  g0[1] = lds_off;                                     // lds_addr (bytes)
  g0[2] = (unsigned)(ga & 0xFFFFFFFFu);                // global_addr[31:0]
  g0[3] = ((unsigned)(ga >> 32) & 0x01FFFFFFu) | 0x80000000u;  // addr[56:32] | type=2
  unsigned td0 = (unsigned)stride;                     // tensor_dim0 (x-OOB bound)
  unsigned td1 = 0x007FFFFFu;                          // tensor_dim1: huge, no OOB
  ivec8 g1;
  g1[0] = 0x00010000;                                  // data_size=1 (2 bytes)
  g1[1] = (int)((td0 & 0xFFFFu) << 16);                // tensor_dim0 lo16
  g1[2] = (int)((td0 >> 16) | ((td1 & 0xFFFFu) << 16));// td0 hi16 | td1 lo16
  g1[3] = (int)((td1 >> 16) | ((unsigned)rowlen << 16));// td1 hi16 | tile_dim0
  g1[4] = rows & 0xFFFF;                               // tile_dim1 (tile_dim2=0)
  g1[5] = stride;                                      // tensor_dim0_stride lo32
  g1[6] = 0;                                           // stride hi16 | td1_stride lo
  g1[7] = 0;
  ivec4 z4 = {0, 0, 0, 0};
  ivec8 z8 = {0, 0, 0, 0, 0, 0, 0, 0};
  __builtin_amdgcn_tensor_load_to_lds(g0, g1, z4, z4, z8, 0);
}
#endif

// ---------------- x = tok_emb[idx] + pos_emb ----------------
__global__ __launch_bounds__(256) void embed_kernel(const int* __restrict__ idx,
                                                    const float* __restrict__ tok,
                                                    const float* __restrict__ pos,
                                                    float* __restrict__ x) {
  long i = (long)blockIdx.x * blockDim.x + threadIdx.x;
  if (i >= (long)MTOK * Dd) return;
  int d  = (int)(i % Dd);
  int bt = (int)(i / Dd);
  int t  = bt % Tt;
  x[i] = tok[(long)idx[bt] * Dd + d] + pos[(long)t * Dd + d];
}

// ---- one-time weight repack: fp32 [K,N] -> bf16 transposed [N,K] ----
// Makes every GEMM B block-tile a contiguous row-major 2-D tile (TDM-friendly,
// no LDS scatter) and halves weight bytes for all subsequent reads.
__global__ __launch_bounds__(256) void cast_transpose_kernel(const float* __restrict__ src,
                                                             u16* __restrict__ dst,
                                                             int K, int N) {
  long total = (long)K * N;
  long stride = (long)gridDim.x * blockDim.x;
  for (long i = (long)blockIdx.x * blockDim.x + threadIdx.x; i < total; i += stride) {
    long n = i / K, k = i % K;                 // output-major: coalesced writes
    dst[i] = f2bf(src[k * (long)N + n]);
  }
}

// ---------------- LayerNorm: fp32 in -> bf16 out (feeds WMMA GEMMs) ----------------
__global__ __launch_bounds__(256) void layernorm_kernel(const float* __restrict__ x,
                                                        const float* __restrict__ g,
                                                        const float* __restrict__ b,
                                                        u16* __restrict__ y) {
  int row = blockIdx.x;
  int tid = threadIdx.x;
  const float* xr = x + (long)row * Dd;
  float lv[4];
  float s = 0.f;
#pragma unroll
  for (int i = 0; i < 4; ++i) { lv[i] = xr[tid + i * 256]; s += lv[i]; }

  __shared__ float red[8];
#pragma unroll
  for (int m = 1; m < 32; m <<= 1) s += __shfl_xor(s, m, 32);
  if ((tid & 31) == 0) red[tid >> 5] = s;
  __syncthreads();
  float tot = 0.f;
#pragma unroll
  for (int w = 0; w < 8; ++w) tot += red[w];
  float mean = tot * (1.f / Dd);
  __syncthreads();

  float v = 0.f;
#pragma unroll
  for (int i = 0; i < 4; ++i) { float d0 = lv[i] - mean; v += d0 * d0; }
#pragma unroll
  for (int m = 1; m < 32; m <<= 1) v += __shfl_xor(v, m, 32);
  if ((tid & 31) == 0) red[tid >> 5] = v;
  __syncthreads();
  float vt = 0.f;
#pragma unroll
  for (int w = 0; w < 8; ++w) vt += red[w];
  float rstd = rsqrtf(vt * (1.f / Dd) + 1e-5f);

#pragma unroll
  for (int i = 0; i < 4; ++i) {
    int c = tid + i * 256;
    y[(long)row * Dd + c] = f2bf((lv[i] - mean) * rstd * g[c] + b[c]);
  }
}

// -------- bf16 WMMA GEMM: out[M,N] = A[M,K] @ Wt[N,K]^T + bias (+res, gelu) --------
// 128x128 block, 8 wave32 waves (4x2), 64-deep K slabs, TDM double-buffered LDS.
#define BM  128
#define BN  128
#define BK2 64

template <bool RES, bool OUTF, bool OUTB, bool ACT>
__global__ __launch_bounds__(256) void gemm_wmma_kernel(
    const u16* __restrict__ A, const u16* __restrict__ Wt,
    const float* __restrict__ bias, const float* __restrict__ residual,
    float* __restrict__ outF, u16* __restrict__ outB,
    int M, int N, int K) {
#ifdef USE_TDM
  __shared__ __align__(16) u16 As[2][BM][BK2];
  __shared__ __align__(16) u16 Bs[2][BN][BK2];
#else
  __shared__ __align__(16) u16 As[1][BM][BK2];
  __shared__ __align__(16) u16 Bs[1][BN][BK2];
#endif

  int tid  = threadIdx.x;
  int lane = tid & 31, wave = tid >> 5;
  int hi = lane >> 4, ln = lane & 15;
  int wm = wave & 3;   // 4 wave-rows of 32
  int wn = wave >> 2;  // 2 wave-cols of 64
  long rowBase = (long)blockIdx.y * BM;
  long colBase = (long)blockIdx.x * BN;

  fvec8 acc[2][4];
#pragma unroll
  for (int i = 0; i < 2; ++i)
#pragma unroll
    for (int j = 0; j < 4; ++j) acc[i][j] = zero8();

  const u16* Abase = A  + rowBase * K;
  const u16* Bbase = Wt + colBase * K;
  int steps = K / BK2;

#ifdef USE_TDM
  if (wave == 0) {  // wave 0 drives the Tensor Data Mover (EXEC-independent DMA)
    tdm_load_2d((unsigned)(size_t)&As[0][0][0], Abase, BM, BK2, K);
    tdm_load_2d((unsigned)(size_t)&Bs[0][0][0], Bbase, BN, BK2, K);
  }
#endif

  for (int it = 0; it < steps; ++it) {
#ifdef USE_TDM
    int buf = it & 1;
    if (wave == 0) {
      if (it + 1 < steps) {   // prefetch slab it+1 into the other buffer
        tdm_load_2d((unsigned)(size_t)&As[buf ^ 1][0][0],
                    Abase + (long)(it + 1) * BK2, BM, BK2, K);
        tdm_load_2d((unsigned)(size_t)&Bs[buf ^ 1][0][0],
                    Bbase + (long)(it + 1) * BK2, BN, BK2, K);
        __builtin_amdgcn_s_wait_tensorcnt(2);  // slab `it` fully in LDS
      } else {
        __builtin_amdgcn_s_wait_tensorcnt(0);
      }
    }
    __syncthreads();
#else
    int buf = 0;
    {  // manual staging fallback: both tiles contiguous, 4x16B per thread each
      int r = tid >> 1;
      int c0 = (tid & 1) * 32;
      const u16* ag = Abase + (long)r * K + (long)it * BK2 + c0;
      const u16* bg = Bbase + (long)r * K + (long)it * BK2 + c0;
#pragma unroll
      for (int i = 0; i < 4; ++i) {
        *(usvec8*)&As[0][r][c0 + i * 8] = *(const usvec8*)(ag + i * 8);
        *(usvec8*)&Bs[0][r][c0 + i * 8] = *(const usvec8*)(bg + i * 8);
      }
      if (it + 1 < steps) {
        __builtin_prefetch(ag + BK2, 0, 1);
        __builtin_prefetch(bg + BK2, 0, 1);
      }
    }
    __syncthreads();
#endif

#pragma unroll
    for (int ks = 0; ks < BK2; ks += 32) {
      AFrag af[2], bfr[4];
#pragma unroll
      for (int sm = 0; sm < 2; ++sm) {
        int m = wm * 32 + sm * 16 + ln;   // A frag: row=lane&15, K-half by lane-hi
        af[sm].h[0] = *(const usvec8*)&As[buf][m][ks + hi * 8];
        af[sm].h[1] = *(const usvec8*)&As[buf][m][ks + 16 + hi * 8];
      }
#pragma unroll
      for (int sn = 0; sn < 4; ++sn) {
        int n = wn * 64 + sn * 16 + ln;   // B frag: col=lane&15, K-half by lane-hi
        bfr[sn].h[0] = *(const usvec8*)&Bs[buf][n][ks + hi * 8];
        bfr[sn].h[1] = *(const usvec8*)&Bs[buf][n][ks + 16 + hi * 8];
      }
#pragma unroll
      for (int sm = 0; sm < 2; ++sm)
#pragma unroll
        for (int sn = 0; sn < 4; ++sn)
          acc[sm][sn] = __builtin_amdgcn_wmma_f32_16x16x32_bf16(
              false, af[sm].v, false, bfr[sn].v, (short)0, acc[sm][sn], false, false);
    }
    __syncthreads();
  }

  // straight-line epilogue (all policy is compile-time)
#pragma unroll
  for (int sm = 0; sm < 2; ++sm) {
#pragma unroll
    for (int sn = 0; sn < 4; ++sn) {
      long col = colBase + wn * 64 + sn * 16 + ln;
      float bv = bias[col];
#pragma unroll
      for (int j = 0; j < 8; ++j) {
        long row = rowBase + wm * 32 + sm * 16 + hi * 8 + j;  // C layout: M = hi*8+j
        float v = acc[sm][sn][j] + bv;
        if constexpr (RES)  v += residual[row * N + col];
        if constexpr (ACT)  v = 0.5f * v * (1.f + erff(v * 0.70710678118654752f));
        if constexpr (OUTF) outF[row * N + col] = v;
        if constexpr (OUTB) outB[row * N + col] = f2bf(v);
      }
    }
  }
}

// ---------------- V [B,T,H,HD] -> Vt [B,H,HD,T] for contiguous PV B-fragments ------
__global__ __launch_bounds__(256) void transpose_v_kernel(const u16* __restrict__ v,
                                                          u16* __restrict__ vT) {
  long i = (long)blockIdx.x * blockDim.x + threadIdx.x;
  if (i >= (long)MTOK * Dd) return;
  int d  = (int)(i % Dd);
  long bt = i / Dd;
  int t = (int)(bt % Tt);
  int b = (int)(bt / Tt);
  int h = d >> 6, c = d & 63;
  vT[(((long)(b * Hh + h) * HD) + c) * Tt + t] = v[i];
}

// ---------------- flash attention: 1 wave per (b,h,16-query tile), 32-key blocks ---
__global__ __launch_bounds__(128) void attn_kernel(const u16* __restrict__ q,
                                                   const u16* __restrict__ k,
                                                   const u16* __restrict__ vT,
                                                   u16* __restrict__ y) {
  __shared__ __align__(16) u16 pbuf[4][16][32];  // per-wave P tile (C-layout -> A-layout)
  int lane = threadIdx.x & 31, wid = threadIdx.x >> 5;
  int hi = lane >> 4, ln = lane & 15;
  int tile = blockIdx.x * 4 + wid;  // (b*H + h)*64 + qt
  int qt = tile & 63;
  int h  = (tile >> 6) & 15;
  int b  = tile >> 10;
  const float scale = 0.125f;  // 1/sqrt(HD)

  AFrag qf[2];
  {
    long qrow = ((long)(b * Tt) + qt * 16 + ln) * Dd + h * HD;
#pragma unroll
    for (int c = 0; c < 2; ++c) {
      qf[c].h[0] = *(const usvec8*)(q + qrow + c * 32 + hi * 8);
      qf[c].h[1] = *(const usvec8*)(q + qrow + c * 32 + 16 + hi * 8);
    }
  }

  float mrow[8], lrow[8];
  fvec8 o[4];
#pragma unroll
  for (int j = 0; j < 8; ++j) { mrow[j] = -1e30f; lrow[j] = 0.f; }
#pragma unroll
  for (int c = 0; c < 4; ++c) o[c] = zero8();

  int qlast = qt * 16 + 15;
  for (int jb = 0; jb <= qlast; jb += 32) {
    // S[16 x 32] = Q(16x64) @ K^T, as two 16x16 C-tiles (key halves)
    fvec8 s01[2];
#pragma unroll
    for (int sh = 0; sh < 2; ++sh) {
      fvec8 sacc = zero8();
      long krow = ((long)(b * Tt) + jb + sh * 16 + ln) * Dd + h * HD;
#pragma unroll
      for (int c = 0; c < 2; ++c) {
        AFrag kf;  // B frag straight from global: contiguous in channel
        kf.h[0] = *(const usvec8*)(k + krow + c * 32 + hi * 8);
        kf.h[1] = *(const usvec8*)(k + krow + c * 32 + 16 + hi * 8);
        sacc = __builtin_amdgcn_wmma_f32_16x16x32_bf16(
            false, qf[c].v, false, kf.v, (short)0, sacc, false, false);
      }
      s01[sh] = sacc;
    }

    // scale + causal mask + online softmax (row = hi*8+j lives on one lane half)
#pragma unroll
    for (int j = 0; j < 8; ++j) {
      int rg = qt * 16 + hi * 8 + j;
      float s0 = s01[0][j] * scale;
      float s1 = s01[1][j] * scale;
      if (jb + ln > rg)      s0 = -1e30f;
      if (jb + 16 + ln > rg) s1 = -1e30f;
      float pm = fmaxf(s0, s1);
#pragma unroll
      for (int m = 1; m < 16; m <<= 1) pm = fmaxf(pm, __shfl_xor(pm, m, 32));
      float mnew  = fmaxf(mrow[j], pm);
      float alpha = __expf(mrow[j] - mnew);
      float p0 = __expf(s0 - mnew);
      float p1 = __expf(s1 - mnew);
      float ps = p0 + p1;
#pragma unroll
      for (int m = 1; m < 16; m <<= 1) ps += __shfl_xor(ps, m, 32);
      lrow[j] = lrow[j] * alpha + ps;
      mrow[j] = mnew;
#pragma unroll
      for (int c = 0; c < 4; ++c) o[c][j] *= alpha;
      int r = hi * 8 + j;
      pbuf[wid][r][ln]      = f2bf(p0);
      pbuf[wid][r][16 + ln] = f2bf(p1);
    }

    // O += P(16x32) @ V(32x64): reload P as an A-fragment from LDS
    AFrag pf;
    pf.h[0] = *(const usvec8*)&pbuf[wid][ln][hi * 8];
    pf.h[1] = *(const usvec8*)&pbuf[wid][ln][16 + hi * 8];
#pragma unroll
    for (int c = 0; c < 4; ++c) {
      AFrag vf;  // Vt is [B,H,HD,T]: contiguous over key index
      long vrow = (((long)(b * Hh + h) * HD) + c * 16 + ln) * Tt + jb + hi * 8;
      vf.h[0] = *(const usvec8*)(vT + vrow);
      vf.h[1] = *(const usvec8*)(vT + vrow + 16);
      o[c] = __builtin_amdgcn_wmma_f32_16x16x32_bf16(
          false, pf.v, false, vf.v, (short)0, o[c], false, false);
    }
  }

#pragma unroll
  for (int c = 0; c < 4; ++c)
#pragma unroll
    for (int j = 0; j < 8; ++j) {
      long t = qt * 16 + hi * 8 + j;
      y[((long)(b * Tt) + t) * Dd + h * HD + c * 16 + ln] = f2bf(o[c][j] / lrow[j]);
    }
}

// ---------------- per-row CE loss; then deterministic single-block mean ------------
__global__ __launch_bounds__(256) void row_loss_kernel(const float* __restrict__ logits,
                                                       const int* __restrict__ tgt,
                                                       float* __restrict__ rl) {
  int row = blockIdx.x, tid = threadIdx.x;
  const float* lr = logits + (long)row * Vv;
  __shared__ float red[8];

  float m = -1e30f;
  for (int i = tid; i < Vv; i += 256) m = fmaxf(m, lr[i]);
#pragma unroll
  for (int msk = 1; msk < 32; msk <<= 1) m = fmaxf(m, __shfl_xor(m, msk, 32));
  if ((tid & 31) == 0) red[tid >> 5] = m;
  __syncthreads();
  float mm = -1e30f;
#pragma unroll
  for (int w = 0; w < 8; ++w) mm = fmaxf(mm, red[w]);
  __syncthreads();

  float s = 0.f;
  for (int i = tid; i < Vv; i += 256) s += __expf(lr[i] - mm);
#pragma unroll
  for (int msk = 1; msk < 32; msk <<= 1) s += __shfl_xor(s, msk, 32);
  if ((tid & 31) == 0) red[tid >> 5] = s;
  __syncthreads();
  if (tid == 0) {
    float ss = 0.f;
#pragma unroll
    for (int w = 0; w < 8; ++w) ss += red[w];
    rl[row] = -(lr[tgt[row]] - mm - __logf(ss));
  }
}

__global__ __launch_bounds__(256) void loss_reduce_kernel(const float* __restrict__ rl,
                                                          float* __restrict__ out, int n) {
  __shared__ float red[8];
  float s = 0.f;
  for (int i = threadIdx.x; i < n; i += 256) s += rl[i];
#pragma unroll
  for (int m = 1; m < 32; m <<= 1) s += __shfl_xor(s, m, 32);
  if ((threadIdx.x & 31) == 0) red[threadIdx.x >> 5] = s;
  __syncthreads();
  if (threadIdx.x == 0) {
    float t = 0.f;
#pragma unroll
    for (int w = 0; w < 8; ++w) t += red[w];
    out[0] = t / n;
  }
}

// ---------------- host-side orchestration ----------------
extern "C" void kernel_launch(void* const* d_in, const int* in_sizes, int n_in,
                              void* d_out, int out_size, void* d_ws, size_t ws_size,
                              hipStream_t stream) {
  (void)in_sizes; (void)n_in; (void)out_size; (void)ws_size;

  const int*   idx     = (const int*)  d_in[0];
  const int*   targets = (const int*)  d_in[1];
  const float* tok_emb = (const float*)d_in[2];
  const float* pos_emb = (const float*)d_in[3];
  const float* ln1_g   = (const float*)d_in[4];
  const float* ln1_b   = (const float*)d_in[5];
  const float* wq      = (const float*)d_in[6];
  const float* bq      = (const float*)d_in[7];
  const float* wk      = (const float*)d_in[8];
  const float* bk      = (const float*)d_in[9];
  const float* wvw     = (const float*)d_in[10];
  const float* bvv     = (const float*)d_in[11];
  const float* ln2_g   = (const float*)d_in[12];
  const float* ln2_b   = (const float*)d_in[13];
  const float* w1      = (const float*)d_in[14];
  const float* b1      = (const float*)d_in[15];
  const float* w2      = (const float*)d_in[16];
  const float* b2      = (const float*)d_in[17];
  const float* lnf_g   = (const float*)d_in[18];
  const float* lnf_b   = (const float*)d_in[19];
  const float* lm_w    = (const float*)d_in[20];
  const float* lm_b    = (const float*)d_in[21];

  float* logits = (float*)d_out;
  float* lossp  = (float*)d_out + (long)MTOK * Vv;

  char* ws = (char*)d_ws;
  size_t off = 0;
  auto alloc = [&](size_t bytes) -> void* {
    void* p = ws + off;
    off = (off + bytes + 255) & ~(size_t)255;
    return p;
  };

  // bf16 transposed weight shadows (~201 MB) + activations (~45 MB)
  u16* wqT = (u16*)alloc((size_t)Ll * Dd * Dd * 2);
  u16* wkT = (u16*)alloc((size_t)Ll * Dd * Dd * 2);
  u16* wvT = (u16*)alloc((size_t)Ll * Dd * Dd * 2);
  u16* w1T = (u16*)alloc((size_t)Ll * Dd * 4 * Dd * 2);
  u16* w2T = (u16*)alloc((size_t)Ll * 4 * Dd * Dd * 2);
  u16* lmT = (u16*)alloc((size_t)Dd * Vv * 2);
  float* x   = (float*)alloc((size_t)MTOK * Dd * 4);
  u16* a_bf  = (u16*)alloc((size_t)MTOK * Dd * 2);
  u16* q_bf  = (u16*)alloc((size_t)MTOK * Dd * 2);
  u16* k_bf  = (u16*)alloc((size_t)MTOK * Dd * 2);
  u16* v_bf  = (u16*)alloc((size_t)MTOK * Dd * 2);
  u16* vT_bf = (u16*)alloc((size_t)MTOK * Dd * 2);
  u16* y_bf  = (u16*)alloc((size_t)MTOK * Dd * 2);
  u16* h1_bf = (u16*)alloc((size_t)MTOK * 4 * Dd * 2);
  float* rl  = (float*)alloc((size_t)MTOK * 4);

  auto castT = [&](const float* s, u16* d, int K, int N) {
    cast_transpose_kernel<<<4096, 256, 0, stream>>>(s, d, K, N);
  };
  for (int l = 0; l < Ll; ++l) {
    castT(wq  + (size_t)l * Dd * Dd,      wqT + (size_t)l * Dd * Dd,      Dd, Dd);
    castT(wk  + (size_t)l * Dd * Dd,      wkT + (size_t)l * Dd * Dd,      Dd, Dd);
    castT(wvw + (size_t)l * Dd * Dd,      wvT + (size_t)l * Dd * Dd,      Dd, Dd);
    castT(w1  + (size_t)l * Dd * 4 * Dd,  w1T + (size_t)l * Dd * 4 * Dd,  Dd, 4 * Dd);
    castT(w2  + (size_t)l * 4 * Dd * Dd,  w2T + (size_t)l * 4 * Dd * Dd,  4 * Dd, Dd);
  }
  castT(lm_w, lmT, Dd, Vv);

  long ntot = (long)MTOK * Dd;
  embed_kernel<<<(unsigned)((ntot + 255) / 256), 256, 0, stream>>>(idx, tok_emb, pos_emb, x);

  for (int l = 0; l < Ll; ++l) {
    layernorm_kernel<<<MTOK, 256, 0, stream>>>(x, ln1_g + (long)l * Dd, ln1_b + (long)l * Dd, a_bf);

    dim3 gDD(Dd / BN, MTOK / BM);
    gemm_wmma_kernel<false, false, true, false><<<gDD, 256, 0, stream>>>(
        a_bf, wqT + (size_t)l * Dd * Dd, bq + (long)l * Dd, nullptr, nullptr, q_bf,
        MTOK, Dd, Dd);
    gemm_wmma_kernel<false, false, true, false><<<gDD, 256, 0, stream>>>(
        a_bf, wkT + (size_t)l * Dd * Dd, bk + (long)l * Dd, nullptr, nullptr, k_bf,
        MTOK, Dd, Dd);
    gemm_wmma_kernel<false, false, true, false><<<gDD, 256, 0, stream>>>(
        a_bf, wvT + (size_t)l * Dd * Dd, bvv + (long)l * Dd, nullptr, nullptr, v_bf,
        MTOK, Dd, Dd);

    transpose_v_kernel<<<(unsigned)((ntot + 255) / 256), 256, 0, stream>>>(v_bf, vT_bf);
    attn_kernel<<<Bb * Hh * (Tt / 16) / 4, 128, 0, stream>>>(q_bf, k_bf, vT_bf, y_bf);

    // output projection reuses wv (as in reference), residual fused, x in place
    gemm_wmma_kernel<true, true, false, false><<<gDD, 256, 0, stream>>>(
        y_bf, wvT + (size_t)l * Dd * Dd, bvv + (long)l * Dd, x, x, nullptr,
        MTOK, Dd, Dd);

    layernorm_kernel<<<MTOK, 256, 0, stream>>>(x, ln2_g + (long)l * Dd, ln2_b + (long)l * Dd, a_bf);

    dim3 gFC1(4 * Dd / BN, MTOK / BM);
    gemm_wmma_kernel<false, false, true, true><<<gFC1, 256, 0, stream>>>(
        a_bf, w1T + (size_t)l * Dd * 4 * Dd, b1 + (long)l * 4 * Dd, nullptr, nullptr,
        h1_bf, MTOK, 4 * Dd, Dd);
    gemm_wmma_kernel<true, true, false, false><<<gDD, 256, 0, stream>>>(
        h1_bf, w2T + (size_t)l * 4 * Dd * Dd, b2 + (long)l * Dd, x, x, nullptr,
        MTOK, Dd, 4 * Dd);
  }

  layernorm_kernel<<<MTOK, 256, 0, stream>>>(x, lnf_g, lnf_b, a_bf);
  dim3 gLM(Vv / BN, MTOK / BM);
  gemm_wmma_kernel<false, true, false, false><<<gLM, 256, 0, stream>>>(
      a_bf, lmT, lm_b, nullptr, logits, nullptr, MTOK, Vv, Dd);

  row_loss_kernel<<<MTOK, 256, 0, stream>>>(logits, targets, rl);
  loss_reduce_kernel<<<1, 256, 0, stream>>>(rl, lossp, MTOK);
}
